// LSTMAttentionLSTM_22256520528893
// MI455X (gfx1250) — matrix operand order repaired and checked by reading, not compiled, not run on hardware
//
#include <hip/hip_runtime.h>

#define B_   128
#define S_   512
#define IN_  128
#define H_   512
#define OUT_ 128
#define T_   64

typedef __attribute__((ext_vector_type(16))) __bf16         v16bf;
typedef __attribute__((ext_vector_type(16))) unsigned short v16u;
typedef __attribute__((ext_vector_type(8)))  unsigned short v8u;
typedef __attribute__((ext_vector_type(8)))  float          v8f;

// ---------- fp32 -> bf16 (round-to-nearest-even) ----------
__device__ __forceinline__ unsigned short f2bf(float f) {
    union { float f; unsigned u; } cv; cv.f = f;
    unsigned u = cv.u;
    u += 0x7FFFu + ((u >> 16) & 1u);
    return (unsigned short)(u >> 16);
}

// ---------- bf16 WMMA fragment loaders (pure b128 loads, no conversion) ----------
// A fragment (16x32, MxK): rowptr = &A[row][0] (bf16, row = lane&15).
// halves 0..15 <- k0 + hi*16 .. +15 (contiguous)
__device__ __forceinline__ v16bf load_a_bf(const unsigned short* __restrict__ rowptr,
                                           int k0, int hi) {
    v8u lo = *(const v8u*)(rowptr + k0 + hi * 16);
    v8u hh = *(const v8u*)(rowptr + k0 + hi * 16 + 8);
    v16u t = __builtin_shufflevector(lo, hh, 0,1,2,3,4,5,6,7,8,9,10,11,12,13,14,15);
    return __builtin_bit_cast(v16bf, t);
}

// B fragment (32x16, KxN), weights N-major (W[n][k]): lane n = lane&15 holds
// 16 contiguous k at k0 + (lane>=16)*16.  p = &W[n][k0 + kh]
__device__ __forceinline__ v16bf load_b_bf(const unsigned short* __restrict__ p) {
    v8u lo = *(const v8u*)(p);
    v8u hh = *(const v8u*)(p + 8);
    v16u t = __builtin_shufflevector(lo, hh, 0,1,2,3,4,5,6,7,8,9,10,11,12,13,14,15);
    return __builtin_bit_cast(v16bf, t);
}

__device__ __forceinline__ v8f wmma_bf16(v16bf a, v16bf b, v8f c) {
    return __builtin_amdgcn_wmma_f32_16x16x32_bf16(false, a, false, b, (short)0, c, false, false);
}

__device__ __forceinline__ float sigmoidf_(float x) { return 1.0f / (1.0f + __expf(-x)); }

// ---------------------------------------------------------------------------
// One-shot fp32 -> bf16 conversion (grid-stride)
// ---------------------------------------------------------------------------
__global__ __launch_bounds__(256) void cvt_kernel(const float* __restrict__ in,
                                                  unsigned short* __restrict__ out, int n) {
    for (int i = blockIdx.x * blockDim.x + threadIdx.x; i < n; i += gridDim.x * blockDim.x)
        out[i] = f2bf(in[i]);
}

__global__ void init_kernel(float* __restrict__ h0, unsigned short* __restrict__ h0b,
                            float* __restrict__ c) {
    const int i = blockIdx.x * blockDim.x + threadIdx.x;
    if (i < B_ * H_) { h0[i] = 0.f; h0b[i] = 0; c[i] = 0.f; }
}

// ---------------------------------------------------------------------------
// Encoder step: gates = x_t @ Wih^T + h @ Whh^T + biases; LSTM cell update.
// 256 waves: (mtile 0..7) x (ntile 0..31); each wave: 16x16 tile of all 4 gates.
// ---------------------------------------------------------------------------
__global__ __launch_bounds__(256) void enc_step_kernel(
    const unsigned short* __restrict__ srcb, const unsigned short* __restrict__ Wih,
    const unsigned short* __restrict__ Whh, const float* __restrict__ bih,
    const float* __restrict__ bhh, const unsigned short* __restrict__ h_in,
    unsigned short* __restrict__ h_outb, float* __restrict__ h_outf,
    float* __restrict__ c, float* __restrict__ enc_out, int t)
{
    const int lane  = threadIdx.x & 31;
    const int wid   = blockIdx.x * 8 + (threadIdx.x >> 5);
    const int mtile = wid >> 5;          // 0..7
    const int ntile = wid & 31;          // 0..31
    const int hi    = lane >> 4;
    const int row   = lane & 15;
    const int kh    = hi * 16;
    const int ncol  = ntile * 16 + row;  // output column within H (lane's N)

    const unsigned short* arow_x = srcb + (size_t)(mtile * 16 + row) * (S_ * IN_) + (size_t)t * IN_;
    const unsigned short* arow_h = h_in + (size_t)(mtile * 16 + row) * H_;

    v8f acc0 = {0,0,0,0,0,0,0,0};
    v8f acc1 = acc0, acc2 = acc0, acc3 = acc0;

    // x @ Wih^T   (K = IN = 128)
    #pragma unroll
    for (int k0 = 0; k0 < IN_; k0 += 32) {
        v16bf a = load_a_bf(arow_x, k0, hi);
        acc0 = wmma_bf16(a, load_b_bf(Wih + (size_t)(0 * H_ + ncol) * IN_ + k0 + kh), acc0);
        acc1 = wmma_bf16(a, load_b_bf(Wih + (size_t)(1 * H_ + ncol) * IN_ + k0 + kh), acc1);
        acc2 = wmma_bf16(a, load_b_bf(Wih + (size_t)(2 * H_ + ncol) * IN_ + k0 + kh), acc2);
        acc3 = wmma_bf16(a, load_b_bf(Wih + (size_t)(3 * H_ + ncol) * IN_ + k0 + kh), acc3);
    }
    // h @ Whh^T   (K = H = 512)
    for (int k0 = 0; k0 < H_; k0 += 32) {
        v16bf a = load_a_bf(arow_h, k0, hi);
        acc0 = wmma_bf16(a, load_b_bf(Whh + (size_t)(0 * H_ + ncol) * H_ + k0 + kh), acc0);
        acc1 = wmma_bf16(a, load_b_bf(Whh + (size_t)(1 * H_ + ncol) * H_ + k0 + kh), acc1);
        acc2 = wmma_bf16(a, load_b_bf(Whh + (size_t)(2 * H_ + ncol) * H_ + k0 + kh), acc2);
        acc3 = wmma_bf16(a, load_b_bf(Whh + (size_t)(3 * H_ + ncol) * H_ + k0 + kh), acc3);
    }

    const float b0 = bih[0 * H_ + ncol] + bhh[0 * H_ + ncol];
    const float b1 = bih[1 * H_ + ncol] + bhh[1 * H_ + ncol];
    const float b2 = bih[2 * H_ + ncol] + bhh[2 * H_ + ncol];
    const float b3 = bih[3 * H_ + ncol] + bhh[3 * H_ + ncol];

    #pragma unroll
    for (int r = 0; r < 8; ++r) {
        const int m = mtile * 16 + r + 8 * hi;     // batch row
        const float gi = sigmoidf_(acc0[r] + b0);
        const float gf = sigmoidf_(acc1[r] + b1);
        const float gg = tanhf(acc2[r] + b2);
        const float go = sigmoidf_(acc3[r] + b3);
        const size_t ci = (size_t)m * H_ + ncol;
        const float cn = gf * c[ci] + gi * gg;
        c[ci] = cn;
        const float hn = go * tanhf(cn);
        h_outf[ci] = hn;
        h_outb[ci] = f2bf(hn);
        enc_out[((size_t)m * S_ + t) * H_ + ncol] = hn;
    }
}

// ---------------------------------------------------------------------------
// Decoder step: x = concat(ctx, zeros) -> only first H cols of dec_Wih matter.
// ---------------------------------------------------------------------------
__global__ __launch_bounds__(256) void dec_step_kernel(
    const unsigned short* __restrict__ ctxb, const unsigned short* __restrict__ Wih,
    const unsigned short* __restrict__ Whh, const float* __restrict__ bih,
    const float* __restrict__ bhh, const unsigned short* __restrict__ h_in,
    unsigned short* __restrict__ h_outb, float* __restrict__ h_outf,
    float* __restrict__ c)
{
    const int lane  = threadIdx.x & 31;
    const int wid   = blockIdx.x * 8 + (threadIdx.x >> 5);
    const int mtile = wid >> 5;
    const int ntile = wid & 31;
    const int hi    = lane >> 4;
    const int row   = lane & 15;
    const int kh    = hi * 16;
    const int ncol  = ntile * 16 + row;
    const int ldWih = H_ + IN_;          // dec_Wih is (4H, H+IN)

    const unsigned short* arow_x = ctxb + (size_t)(mtile * 16 + row) * H_;
    const unsigned short* arow_h = h_in + (size_t)(mtile * 16 + row) * H_;

    v8f acc0 = {0,0,0,0,0,0,0,0};
    v8f acc1 = acc0, acc2 = acc0, acc3 = acc0;

    for (int k0 = 0; k0 < H_; k0 += 32) {
        v16bf a = load_a_bf(arow_x, k0, hi);
        acc0 = wmma_bf16(a, load_b_bf(Wih + (size_t)(0 * H_ + ncol) * ldWih + k0 + kh), acc0);
        acc1 = wmma_bf16(a, load_b_bf(Wih + (size_t)(1 * H_ + ncol) * ldWih + k0 + kh), acc1);
        acc2 = wmma_bf16(a, load_b_bf(Wih + (size_t)(2 * H_ + ncol) * ldWih + k0 + kh), acc2);
        acc3 = wmma_bf16(a, load_b_bf(Wih + (size_t)(3 * H_ + ncol) * ldWih + k0 + kh), acc3);
    }
    for (int k0 = 0; k0 < H_; k0 += 32) {
        v16bf a = load_a_bf(arow_h, k0, hi);
        acc0 = wmma_bf16(a, load_b_bf(Whh + (size_t)(0 * H_ + ncol) * H_ + k0 + kh), acc0);
        acc1 = wmma_bf16(a, load_b_bf(Whh + (size_t)(1 * H_ + ncol) * H_ + k0 + kh), acc1);
        acc2 = wmma_bf16(a, load_b_bf(Whh + (size_t)(2 * H_ + ncol) * H_ + k0 + kh), acc2);
        acc3 = wmma_bf16(a, load_b_bf(Whh + (size_t)(3 * H_ + ncol) * H_ + k0 + kh), acc3);
    }

    const float b0 = bih[0 * H_ + ncol] + bhh[0 * H_ + ncol];
    const float b1 = bih[1 * H_ + ncol] + bhh[1 * H_ + ncol];
    const float b2 = bih[2 * H_ + ncol] + bhh[2 * H_ + ncol];
    const float b3 = bih[3 * H_ + ncol] + bhh[3 * H_ + ncol];

    #pragma unroll
    for (int r = 0; r < 8; ++r) {
        const int m = mtile * 16 + r + 8 * hi;
        const float gi = sigmoidf_(acc0[r] + b0);
        const float gf = sigmoidf_(acc1[r] + b1);
        const float gg = tanhf(acc2[r] + b2);
        const float go = sigmoidf_(acc3[r] + b3);
        const size_t ci = (size_t)m * H_ + ncol;
        const float cn = gf * c[ci] + gi * gg;
        c[ci] = cn;
        const float hn = go * tanhf(cn);
        h_outf[ci] = hn;
        h_outb[ci] = f2bf(hn);
    }
}

// ---------------------------------------------------------------------------
// FC head: pred[b][o] = h @ fc_W^T + fc_b -> out[b][t][o].  64 waves.
// ---------------------------------------------------------------------------
__global__ __launch_bounds__(256) void fc_kernel(
    const unsigned short* __restrict__ h, const unsigned short* __restrict__ fcW,
    const float* __restrict__ fcb, float* __restrict__ out, int t)
{
    const int lane  = threadIdx.x & 31;
    const int wid   = blockIdx.x * 8 + (threadIdx.x >> 5);
    const int mtile = wid >> 3;          // 0..7
    const int ntile = wid & 7;           // 0..7
    const int hi    = lane >> 4;
    const int row   = lane & 15;
    const int kh    = hi * 16;
    const int ncol  = ntile * 16 + row;

    const unsigned short* arow = h + (size_t)(mtile * 16 + row) * H_;
    v8f acc = {0,0,0,0,0,0,0,0};

    for (int k0 = 0; k0 < H_; k0 += 32) {
        v16bf a = load_a_bf(arow, k0, hi);
        acc = wmma_bf16(a, load_b_bf(fcW + (size_t)ncol * H_ + k0 + kh), acc);
    }
    const float bb = fcb[ncol];
    #pragma unroll
    for (int r = 0; r < 8; ++r) {
        const int m = mtile * 16 + r + 8 * hi;
        out[(size_t)m * (T_ * OUT_) + (size_t)t * OUT_ + ncol] = acc[r] + bb;
    }
}

// ---------------------------------------------------------------------------
// Step-invariant attention term: edot[b][s] = enc_out[b][s][:] . attn_W[H:2H]
// ---------------------------------------------------------------------------
__global__ __launch_bounds__(256) void edot_kernel(
    const float* __restrict__ enc_out, const float* __restrict__ attn_W,
    float* __restrict__ edot)
{
    const int lane = threadIdx.x & 31;
    const int wid  = blockIdx.x * 8 + (threadIdx.x >> 5);   // 0..B*S-1
    const float* rowp = enc_out + (size_t)wid * H_;
    float acc = 0.f;
    for (int i = lane; i < H_; i += 32) acc += rowp[i] * attn_W[H_ + i];
    #pragma unroll
    for (int off = 16; off > 0; off >>= 1) acc += __shfl_xor(acc, off, 32);
    if (lane == 0) edot[wid] = acc;
}

// ---------------------------------------------------------------------------
// Per-step attention + context: one block per batch row; ctx written as bf16.
// ---------------------------------------------------------------------------
__global__ __launch_bounds__(256) void attn_ctx_kernel(
    const float* __restrict__ h, const float* __restrict__ enc_out,
    const float* __restrict__ edot, const float* __restrict__ attn_W,
    const float* __restrict__ attn_b, unsigned short* __restrict__ ctxb)
{
    const int b   = blockIdx.x;
    const int tid = threadIdx.x;
    __shared__ float sw[S_];
    __shared__ float red[256];

    // hdot = h[b] . attn_W[:H]
    float part = 0.f;
    for (int i = tid; i < H_; i += 256) part += h[(size_t)b * H_ + i] * attn_W[i];
    red[tid] = part; __syncthreads();
    for (int s = 128; s > 0; s >>= 1) { if (tid < s) red[tid] += red[tid + s]; __syncthreads(); }
    const float hdot = red[0] + attn_b[0];
    __syncthreads();

    // unnormalized softmax weights (energy in (-1,1): no max-shift needed)
    float lsum = 0.f;
    for (int s = tid; s < S_; s += 256) {
        const float w = __expf(tanhf(hdot + edot[(size_t)b * S_ + s]));
        sw[s] = w; lsum += w;
    }
    red[tid] = lsum; __syncthreads();
    for (int s = 128; s > 0; s >>= 1) { if (tid < s) red[tid] += red[tid + s]; __syncthreads(); }
    const float inv = 1.0f / red[0];
    __syncthreads();

    // ctx[b][hc] = (sum_s sw[s] * enc_out[b][s][hc]) * inv   (enc_out is L2-resident)
    for (int hc = tid; hc < H_; hc += 256) {
        float acc = 0.f;
        const float* ep = enc_out + (size_t)b * S_ * H_ + hc;
        for (int s = 0; s < S_; ++s) acc += sw[s] * ep[(size_t)s * H_];
        ctxb[(size_t)b * H_ + hc] = f2bf(acc * inv);
    }
}

// ---------------------------------------------------------------------------
extern "C" void kernel_launch(void* const* d_in, const int* in_sizes, int n_in,
                              void* d_out, int out_size, void* d_ws, size_t ws_size,
                              hipStream_t stream) {
    (void)in_sizes; (void)n_in; (void)out_size; (void)ws_size;

    const float* src  = (const float*)d_in[0];
    const float* eWih = (const float*)d_in[1];
    const float* eWhh = (const float*)d_in[2];
    const float* ebih = (const float*)d_in[3];
    const float* ebhh = (const float*)d_in[4];
    const float* aW   = (const float*)d_in[5];
    const float* ab   = (const float*)d_in[6];
    const float* dWih = (const float*)d_in[7];
    const float* dWhh = (const float*)d_in[8];
    const float* dbih = (const float*)d_in[9];
    const float* dbhh = (const float*)d_in[10];
    const float* fcW  = (const float*)d_in[11];
    const float* fcb  = (const float*)d_in[12];
    float* out = (float*)d_out;

    // ---- fp32 workspace region ----
    float* f       = (float*)d_ws;
    float* hbF0    = f;                                    // B*H
    float* hbF1    = hbF0 + B_ * H_;                       // B*H
    float* cbuf    = hbF1 + B_ * H_;                       // B*H
    float* enc_out = cbuf + B_ * H_;                       // B*S*H (134 MB; L2-resident)
    float* edot    = enc_out + (size_t)B_ * S_ * H_;       // B*S
    // ---- bf16 (ushort) workspace region (16B-aligned: all counts mult. of 8) ----
    unsigned short* u       = (unsigned short*)(edot + (size_t)B_ * S_);
    unsigned short* src_bf  = u;                            // B*S*IN
    unsigned short* eWih_bf = src_bf  + (size_t)B_ * S_ * IN_;   // 4H*IN
    unsigned short* eWhh_bf = eWih_bf + (size_t)4 * H_ * IN_;    // 4H*H
    unsigned short* dWih_bf = eWhh_bf + (size_t)4 * H_ * H_;     // 4H*(H+IN)
    unsigned short* dWhh_bf = dWih_bf + (size_t)4 * H_ * (H_ + IN_); // 4H*H
    unsigned short* fcW_bf  = dWhh_bf + (size_t)4 * H_ * H_;     // OUT*H
    unsigned short* hbB0    = fcW_bf  + (size_t)OUT_ * H_;       // B*H
    unsigned short* hbB1    = hbB0 + B_ * H_;                    // B*H
    unsigned short* ctx_bf  = hbB1 + B_ * H_;                    // B*H

    float* hbF[2]          = { hbF0, hbF1 };
    unsigned short* hbB[2] = { hbB0, hbB1 };

    // ---- one-shot conversions (step-invariant operands -> bf16) ----
    cvt_kernel<<<1024, 256, 0, stream>>>(src,  src_bf,  B_ * S_ * IN_);
    cvt_kernel<<<256,  256, 0, stream>>>(eWih, eWih_bf, 4 * H_ * IN_);
    cvt_kernel<<<512,  256, 0, stream>>>(eWhh, eWhh_bf, 4 * H_ * H_);
    cvt_kernel<<<512,  256, 0, stream>>>(dWih, dWih_bf, 4 * H_ * (H_ + IN_));
    cvt_kernel<<<512,  256, 0, stream>>>(dWhh, dWhh_bf, 4 * H_ * H_);
    cvt_kernel<<<64,   256, 0, stream>>>(fcW,  fcW_bf,  OUT_ * H_);

    init_kernel<<<(B_ * H_ + 255) / 256, 256, 0, stream>>>(hbF0, hbB0, cbuf);

    // ---- encoder: 512 dependent steps, one fused kernel each ----
    for (int t = 0; t < S_; ++t)
        enc_step_kernel<<<32, 256, 0, stream>>>(src_bf, eWih_bf, eWhh_bf, ebih, ebhh,
                                                hbB[t & 1], hbB[(t + 1) & 1],
                                                hbF[(t + 1) & 1], cbuf, enc_out, t);

    // ---- step-invariant attention projection ----
    edot_kernel<<<(B_ * S_) / 8, 256, 0, stream>>>(enc_out, aW, edot);

    // ---- decoder: 64 dependent steps ----
    for (int d = 0; d < T_; ++d) {
        attn_ctx_kernel<<<B_, 256, 0, stream>>>(hbF[d & 1], enc_out, edot, aW, ab, ctx_bf);
        dec_step_kernel<<<32, 256, 0, stream>>>(ctx_bf, dWih_bf, dWhh_bf, dbih, dbhh,
                                                hbB[d & 1], hbB[(d + 1) & 1],
                                                hbF[(d + 1) & 1], cbuf);
        fc_kernel<<<8, 256, 0, stream>>>(hbB[(d + 1) & 1], fcW_bf, fcb, out, d);
    }
}
